// ChapterHead_80874234183882
// MI455X (gfx1250) — matrix operand order repaired and checked by reading, not compiled, not run on hardware
//
#include <hip/hip_runtime.h>

// ---------------------------------------------------------------------------
// ChapterHead forward for MI455X (gfx1250, wave32, WMMA).
// Weights are pre-packed once per launch into bf16 32x128 tiles in workspace:
// the vis_head weight set drops from 250MB f32 (L2-thrashing) to 125MB bf16
// (fully L2-resident at 192MB), and the GEMM hot loop stages tiles with the
// gfx1250 async global->LDS engine, double-buffered via s_wait_asynccnt, so
// tile DMA overlaps v_wmma_f32_16x16x32_bf16 issue.  16-row activation panels
// stay resident in LDS with fused bias+LayerNorm+ReLU epilogues; vis_head
// (274 GFLOP, ~95% of work) is fully fused per (b,c): 768->2048->1024->256.
// Tiles are zero-padded at pack time so the WMMA loop runs unguarded with
// EXEC all-1s (ISA 7.12 requirement) and no per-step exec-mask overhead.
// ---------------------------------------------------------------------------

typedef __attribute__((ext_vector_type(16))) __bf16       v16bf;
typedef __attribute__((ext_vector_type(8)))  float        v8f;
typedef __attribute__((ext_vector_type(8)))  unsigned int v8u;
typedef __attribute__((ext_vector_type(4)))  unsigned int v4u;
typedef __attribute__((ext_vector_type(4)))  int          v4i;

// address-space qualified pointer types
typedef const float        __attribute__((address_space(1)))* gfp;
typedef float              __attribute__((address_space(1)))* gfw;
typedef const unsigned int __attribute__((address_space(1)))* gup;
typedef unsigned int       __attribute__((address_space(1)))* guw;
typedef const v4u          __attribute__((address_space(1)))* gv4up;
typedef __bf16             __attribute__((address_space(3)))* lbf;
typedef unsigned int       __attribute__((address_space(3)))* lup;
typedef float              __attribute__((address_space(3)))* lfp;
typedef char               __attribute__((address_space(3)))* lch;

#define GP(x)  ((gfp)(x))
#define GW(x)  ((gfw)(x))
#define GU(x)  ((gup)(x))
#define LDS_V4U(p)  ((v4u __attribute__((address_space(3)))*)(p))
#define LDS_CV4U(p) ((const v4u __attribute__((address_space(3)))*)(p))

#define TB 256   // 8 waves of 32

// Async global->LDS (signature confirmed by this toolchain's diagnostics:
//  (v4i addrspace(1)*, v4i addrspace(3)*, imm offset, imm cpol))
#if defined(__HIP_DEVICE_COMPILE__) && defined(__has_builtin)
#  if __has_builtin(__builtin_amdgcn_global_load_async_to_lds_b128)
#    define USE_ASYNC_COPY 1
#  endif
#endif
#ifndef USE_ASYNC_COPY
#  define USE_ASYNC_COPY 0
#endif

#if USE_ASYNC_COPY
typedef __attribute__((address_space(1))) v4i gv4i;
typedef __attribute__((address_space(3))) v4i lv4i;
#  if __has_builtin(__builtin_amdgcn_s_wait_asynccnt)
#    define WAIT_ASYNC(n) __builtin_amdgcn_s_wait_asynccnt(n)
#  else
#    define WAIT_ASYNC(n) asm volatile("s_wait_asynccnt %0" ::"i"(n) : "memory")
#  endif
#else
#  define WAIT_ASYNC(n) ((void)0)
#endif

__device__ __forceinline__ unsigned int pack_bf16(float lo, float hi) {
  unsigned int a = __builtin_bit_cast(unsigned int, lo);
  unsigned int b = __builtin_bit_cast(unsigned int, hi);
  a += 0x7FFFu + ((a >> 16) & 1u);   // round-to-nearest-even
  b += 0x7FFFu + ((b >> 16) & 1u);
  return (a >> 16) | (b & 0xFFFF0000u);
}
__device__ __forceinline__ __bf16 f2bf(float f) {
  unsigned int a = __builtin_bit_cast(unsigned int, f);
  a += 0x7FFFu + ((a >> 16) & 1u);
  return __builtin_bit_cast(__bf16, (unsigned short)(a >> 16));
}
__device__ __forceinline__ float bf2f(__bf16 h) { return (float)h; }

// per-clip stride (in uints) of a packed [K,N] weight matrix
__device__ __host__ inline long wstride(int K, int N) {
  return (long)(K >> 5) * ((N + 127) >> 7) * 2048;
}

// copy one 8KB weight tile (2048 uints) global->LDS; 2 b128 chunks per thread
__device__ __forceinline__ void copy_tile(gup src, lup dst, int tid) {
#if USE_ASYNC_COPY
  __builtin_amdgcn_global_load_async_to_lds_b128(
      (gv4i*)(src + tid * 4), (lv4i*)(dst + tid * 4), 0, 0);
  __builtin_amdgcn_global_load_async_to_lds_b128(
      (gv4i*)(src + (tid + 256) * 4), (lv4i*)(dst + (tid + 256) * 4), 0, 0);
#else
  *LDS_V4U(dst + tid * 4)         = *((gv4up)(src + tid * 4));
  *LDS_V4U(dst + (tid + 256) * 4) = *((gv4up)(src + (tid + 256) * 4));
#endif
}

// ---------------------------------------------------------------------------
// WMMA stage: actY[16,N](LDS bf16) = epi(actA[16,K](LDS bf16) @ Wt + bias).
// Wt = pre-packed bf16 tiles (32x128, column-major K-pairs, zero-padded).
// 8 waves; each wave owns one 16-col strip of a 128-col n-block; tiles are
// double-buffered via async-to-LDS so DMA overlaps the WMMA stream.
// ---------------------------------------------------------------------------
template <bool DO_LN, bool DO_RELU, bool DO_OUT>
__device__ __forceinline__ void stage_gemm(
    lbf actA, lbf actY, lup wt0, lup wt1, lfp red,
    gup Wt, gfp bias, gfp gamma, gfp beta,
    int K, int N, gfw outG, int outStride, int outRows)
{
  const int tid  = threadIdx.x;
  const int lane = tid & 31;
  const int wv   = tid >> 5;
  const int col  = lane & 15;
  const int hf   = lane >> 4;        // lane half (wave32)
  const int m0   = hf * 8;           // D rows: r + m0
  const int Kp   = K >> 1;           // bf16 pairs per A row
  const int ksteps = K >> 5;
  const lup actAp = (lup)actA;

  float rs[8], rq[8];
#pragma unroll
  for (int r = 0; r < 8; ++r) { rs[r] = 0.f; rq[r] = 0.f; }

  const int nblocks = (N + 127) >> 7;
  for (int nb = 0; nb < nblocks; ++nb) {
    const int nbase = nb << 7;
    const gup tbase = Wt + (long)nb * ksteps * 2048;
    v8f acc = {0.f, 0.f, 0.f, 0.f, 0.f, 0.f, 0.f, 0.f};

    copy_tile(tbase, wt0, tid);                      // prologue: tile 0
    for (int ks = 0; ks < ksteps; ++ks) {
      const lup cur  = (ks & 1) ? wt1 : wt0;
      const lup nxt  = (ks & 1) ? wt0 : wt1;
      const bool more = (ks + 1 < ksteps);
      if (more) copy_tile(tbase + (long)(ks + 1) * 2048, nxt, tid);
#if USE_ASYNC_COPY
      if (more) WAIT_ASYNC(2);   // current tile landed; next still in flight
      else      WAIT_ASYNC(0);
#endif
      __syncthreads();

      // Unguarded: all waves run WMMA with EXEC all-1s; out-of-range columns
      // hit the zero-padded tile and contribute nothing.
      // A fragment (16x32 bf16): lane row = lane%16;
      // lanes 0-15 take K pairs {0..3, 8..11}, lanes 16-31 {4..7, 12..15}.
      const lup ar = actAp + (lane & 15) * Kp + (ks << 4) + (hf ? 4 : 0);
      const v4u a0 = *LDS_CV4U(ar);
      const v4u a1 = *LDS_CV4U(ar + 8);
      // B fragment (32x16 bf16): lane col = lane%16; same K pairs as A.
      const lup bc = cur + (((wv << 4) + col) << 4) + (hf ? 4 : 0);
      const v4u b0 = *LDS_CV4U(bc);
      const v4u b1 = *LDS_CV4U(bc + 8);
      const v8u au = __builtin_shufflevector(a0, a1, 0, 1, 2, 3, 4, 5, 6, 7);
      const v8u bu = __builtin_shufflevector(b0, b1, 0, 1, 2, 3, 4, 5, 6, 7);

      acc = __builtin_amdgcn_wmma_f32_16x16x32_bf16(
          false, __builtin_bit_cast(v16bf, au),
          false, __builtin_bit_cast(v16bf, bu),
          (short)0, acc, false, false);

      __syncthreads();   // protect nxt buffer from next iteration's DMA
    }

    // Epilogue: bias add, bf16 store of pre-LN values, LN statistics.
    const int  n     = nbase + (wv << 4) + col;
    const bool valid = (n < N);
    float bv = 0.f;
    if (valid) bv = bias[n];
    if (valid) {
#pragma unroll
      for (int r = 0; r < 8; ++r)
        actY[(r + m0) * N + n] = f2bf(acc[r] + bv);
    }
    if constexpr (DO_LN) {
#pragma unroll
      for (int r = 0; r < 8; ++r) {
        float s = valid ? (acc[r] + bv) : 0.f;
        float q = s * s;
#pragma unroll
        for (int xm = 1; xm < 16; xm <<= 1) {
          s += __shfl_xor(s, xm);
          q += __shfl_xor(q, xm);
        }
        rs[r] += s; rq[r] += q;
      }
    }
  }
  __syncthreads();

  if constexpr (DO_LN) {
    if ((lane & 15) == 0) {
#pragma unroll
      for (int r = 0; r < 8; ++r) {
        const int row = m0 + r;
        red[(wv * 16 + row) * 2 + 0] = rs[r];
        red[(wv * 16 + row) * 2 + 1] = rq[r];
      }
    }
    __syncthreads();
    if (tid < 16) {
      float s = 0.f, q = 0.f;
      for (int w2 = 0; w2 < 8; ++w2) {
        s += red[(w2 * 16 + tid) * 2 + 0];
        q += red[(w2 * 16 + tid) * 2 + 1];
      }
      const float mean = s / (float)N;
      const float var  = q / (float)N - mean * mean;
      red[256 + tid * 2 + 0] = mean;
      red[256 + tid * 2 + 1] = rsqrtf(var + 1e-5f);
    }
    __syncthreads();
    for (int idx = tid; idx < (N << 4); idx += TB) {
      const int m  = idx / N;
      const int nn = idx - m * N;
      float v = bf2f(actY[idx]);
      v = (v - red[256 + m * 2]) * red[256 + m * 2 + 1];
      v = v * gamma[nn] + beta[nn];
      if constexpr (DO_RELU) v = fmaxf(v, 0.f);
      actY[idx] = f2bf(v);
    }
    __syncthreads();
  } else if constexpr (DO_RELU) {
    for (int idx = tid; idx < (N << 4); idx += TB)
      actY[idx] = f2bf(fmaxf(bf2f(actY[idx]), 0.f));
    __syncthreads();
  }

  if constexpr (DO_OUT) {
    for (int idx = tid; idx < outRows * N; idx += TB) {
      const int m  = idx / N;
      const int nn = idx - m * N;
      outG[(long)m * outStride + nn] = bf2f(actY[m * N + nn]);
    }
    __syncthreads();
  }
}

// f32 global rows -> bf16 LDS panel; ds_store_b128 per thread-iteration.
__device__ __forceinline__ void load_act(gfp src, long rowStride, int K, lbf dst) {
  const int K8 = K >> 3;
  lup dp = (lup)dst;
  for (int idx = threadIdx.x; idx < 16 * K8; idx += TB) {
    const int  m  = idx / K8;
    const int  j  = idx - m * K8;
    const gfp  sp = src + (long)m * rowStride + (j << 3);
    v4u w;
#pragma unroll
    for (int i = 0; i < 4; ++i) w[i] = pack_bf16(sp[2 * i], sp[2 * i + 1]);
    *LDS_V4U(dp + (m * K8 + j) * 4) = w;
  }
  __syncthreads();
}

// ---------------------------------------------------------------------------
// Weight pre-pack: f32 [C?,K,N] -> bf16 32x128 tiles (column-major K-pairs,
// zero-padded at N edges).  One block packs one tile.
// ---------------------------------------------------------------------------
__global__ void pack_weight_kernel(const float* Wf_, unsigned int* dst_, int K, int N)
{
  const int ksteps  = K >> 5;
  const int nblocks = (N + 127) >> 7;
  int t = blockIdx.x;
  const int ks = t % ksteps;  t /= ksteps;
  const int nb = t % nblocks; const int c = t / nblocks;
  const gfp W  = GP(Wf_) + (long)c * K * N;
  guw dst = ((guw)dst_) + (((long)c * nblocks + nb) * ksteps + ks) * 2048;
  const int nbase = nb << 7;
  const int k0    = ks << 5;
  for (int p = threadIdx.x; p < 2048; p += TB) {
    const int cc = p >> 4, kp = p & 15;
    const int n  = nbase + cc;
    unsigned int w = 0u;
    if (n < N) {
      const long k = k0 + (kp << 1);
      w = pack_bf16(W[k * N + n], W[(k + 1) * N + n]);
    }
    dst[p] = w;
  }
}

// ---------------------------------------------------------------------------
// Kernels
// ---------------------------------------------------------------------------
struct OutProjP {
  const unsigned int *W1,*W2,*W3,*W4;
  const float *b1,*g1,*be1,*b2,*g2,*be2,*b3,*g3,*be3,*b4;
};
struct ClfP {
  const unsigned int *W1,*W2,*W3,*W4,*W5;
  const float *b1,*g1,*be1,*b2,*g2,*be2,*b3,*g3,*be3,*b4,*g4,*be4,*b5;
};
struct WsaAttnP {
  const float *pW,*pb,*pg,*pbeta,*ng,*nb,*Wq,*bq,*Wk,*bk,*Wv,*bv,*bias;
};

// LDS carve offsets (bytes): wt0 0 | wt1 8192 | red 16384 | bufs 17664
#define OFF_WT1 8192
#define OFF_RED 16384
#define OFF_BUF 17664

// lang head: 16 batches x clip c; [16,768] -> LN/ReLU 384 -> ReLU 256
__global__ void lang_head_kernel(const float* lang,
                                 const unsigned int* W1t, const float* b1,
                                 const float* g1, const float* be1,
                                 const unsigned int* W2t, const float* b2,
                                 float* lang_feat)
{
  extern __shared__ char smem[];
  lch base = (lch)smem;
  lup wt0 = (lup)base;
  lup wt1 = (lup)(base + OFF_WT1);
  lfp red = (lfp)(base + OFF_RED);
  lbf bufA = (lbf)(base + OFF_BUF);                 // 16*768*2
  lbf bufB = (lbf)(base + OFF_BUF + 16 * 768 * 2);
  const int c  = blockIdx.x % 17;
  const int b0 = (blockIdx.x / 17) * 16;
  load_act(GP(lang + ((long)b0 * 17 + c) * 768), 17L * 768, 768, bufA);
  stage_gemm<true, true, false>(bufA, bufB, wt0, wt1, red,
      GU(W1t) + c * wstride(768, 384), GP(b1 + c * 384), GP(g1 + c * 384), GP(be1 + c * 384),
      768, 384, GW(nullptr), 0, 0);
  stage_gemm<false, true, true>(bufB, bufA, wt0, wt1, red,
      GU(W2t) + c * wstride(384, 256), GP(b2 + c * 256), GP(nullptr), GP(nullptr),
      384, 256, GW(lang_feat + ((long)b0 * 17 + c) * 256), 17 * 256, 16);
}

// vis head, fully fused per (b,c): [16,768]->2048->1024->256->pool.
__global__ void vis_head_kernel(const float* vis,
                                const unsigned int* W1t, const float* b1,
                                const float* g1, const float* be1,
                                const unsigned int* W2t, const float* b2,
                                const float* g2, const float* be2,
                                const unsigned int* W3t, const float* b3,
                                float* vis_feat)
{
  extern __shared__ char smem[];
  lch base = (lch)smem;
  lup wt0 = (lup)base;
  lup wt1 = (lup)(base + OFF_WT1);
  lfp red = (lfp)(base + OFF_RED);
  lbf bufA = (lbf)(base + OFF_BUF);                  // 16*2048*2
  lbf bufB = (lbf)(base + OFF_BUF + 16 * 2048 * 2);
  const int b = blockIdx.x / 17;
  const int c = blockIdx.x % 17;
  load_act(GP(vis + ((long)b * 17 + c) * 16 * 768), 768, 768, bufA);
  stage_gemm<true, true, false>(bufA, bufB, wt0, wt1, red,
      GU(W1t) + c * wstride(768, 2048), GP(b1 + c * 2048), GP(g1 + c * 2048), GP(be1 + c * 2048),
      768, 2048, GW(nullptr), 0, 0);
  stage_gemm<true, true, false>(bufB, bufA, wt0, wt1, red,
      GU(W2t) + c * wstride(2048, 1024), GP(b2 + c * 1024), GP(g2 + c * 1024), GP(be2 + c * 1024),
      2048, 1024, GW(nullptr), 0, 0);
  stage_gemm<false, false, false>(bufA, bufB, wt0, wt1, red,
      GU(W3t) + c * wstride(1024, 256), GP(b3 + c * 256), GP(nullptr), GP(nullptr),
      1024, 256, GW(nullptr), 0, 0);
  // mean over the 16 SEG rows, then ReLU
  const int n = threadIdx.x;
  if (n < 256) {
    float s = 0.f;
    for (int m = 0; m < 16; ++m) s += bf2f(bufB[m * 256 + n]);
    GW(vis_feat)[((long)b * 17 + c) * 256 + n] = fmaxf(s * (1.f / 16.f), 0.f);
  }
}

// window self-attention core (S=17, NH=16, hd=16).  <1% of FLOPs -> VALU.
__global__ void wsa_attn_kernel(const float* feat, WsaAttnP pp, float* ctx_)
{
  extern __shared__ char smem[];
  lfp x = (lfp)smem;             // 17*256
  lfp q = x + 17 * 256;
  lfp k = q + 17 * 256;
  lfp v = k + 17 * 256;
  const int b    = blockIdx.x;
  const int tid  = threadIdx.x;
  const int lane = tid & 31;
  const int wv   = tid >> 5;
  const float* src = feat + (long)b * 17 * 256;
  const gfp pW = GP(pp.pW), pb = GP(pp.pb), pg = GP(pp.pg), pbeta = GP(pp.pbeta),
            ng = GP(pp.ng), nb = GP(pp.nb), Wq = GP(pp.Wq), bq = GP(pp.bq),
            Wk = GP(pp.Wk), bk = GP(pp.bk), Wv = GP(pp.Wv), bv = GP(pp.bv),
            bias = GP(pp.bias);
  gfw ctx = GW(ctx_);

#if USE_ASYNC_COPY
  for (int ch = tid; ch < (17 * 256) / 4; ch += TB) {
    gv4i* g = (gv4i*)(src + ch * 4);
    lv4i* l = (lv4i*)(x + ch * 4);
    __builtin_amdgcn_global_load_async_to_lds_b128(g, l, 0, 0);
  }
  WAIT_ASYNC(0);
#else
  for (int i = tid; i < 17 * 256; i += TB) x[i] = GP(src)[i];
#endif
  __syncthreads();

  for (int s = wv; s < 17; s += 8) {
    const float pos = ((float)s - 8.0f) / (8.0f + 1e-6f);
    float pe[8]; float s1 = 0.f, s2 = 0.f;
#pragma unroll
    for (int j = 0; j < 8; ++j) {
      const int h = lane + 32 * j;
      const float t = pos * pW[h] + pb[h];
      pe[j] = t; s1 += t; s2 += t * t;
    }
    for (int xm = 1; xm < 32; xm <<= 1) { s1 += __shfl_xor(s1, xm); s2 += __shfl_xor(s2, xm); }
    float mean = s1 / 256.f;
    float rstd = rsqrtf(s2 / 256.f - mean * mean + 1e-5f);
    float xv[8]; s1 = 0.f; s2 = 0.f;
#pragma unroll
    for (int j = 0; j < 8; ++j) {
      const int h = lane + 32 * j;
      const float t = x[s * 256 + h] + (pe[j] - mean) * rstd * pg[h] + pbeta[h];
      xv[j] = t; s1 += t; s2 += t * t;
    }
    for (int xm = 1; xm < 32; xm <<= 1) { s1 += __shfl_xor(s1, xm); s2 += __shfl_xor(s2, xm); }
    mean = s1 / 256.f;
    rstd = rsqrtf(s2 / 256.f - mean * mean + 1e-5f);
#pragma unroll
    for (int j = 0; j < 8; ++j) {
      const int h = lane + 32 * j;
      x[s * 256 + h] = (xv[j] - mean) * rstd * ng[h] + nb[h];
    }
  }
  __syncthreads();

  for (int idx = tid; idx < 17 * 256; idx += TB) {
    const int s = idx >> 8, h = idx & 255;
    float aq = bq[h], ak = bk[h], av = bv[h];
    const lfp xr = x + (s << 8);
    for (int kk = 0; kk < 256; ++kk) {
      const float xvv = xr[kk];
      aq += xvv * Wq[(kk << 8) + h];
      ak += xvv * Wk[(kk << 8) + h];
      av += xvv * Wv[(kk << 8) + h];
    }
    q[idx] = aq; k[idx] = ak; v[idx] = av;
  }
  __syncthreads();

  for (int idx = tid; idx < 16 * 17; idx += TB) {
    const int hh = idx / 17, qp = idx % 17;
    float sc[17]; float mx = -1e30f;
    for (int kp = 0; kp < 17; ++kp) {
      float d = 0.f;
      for (int e = 0; e < 16; ++e)
        d += q[(qp << 8) + hh * 16 + e] * k[(kp << 8) + hh * 16 + e];
      d = d * 0.25f + bias[(hh * 17 + qp) * 17 + kp];
      sc[kp] = d; mx = fmaxf(mx, d);
    }
    float den = 0.f;
    for (int kp = 0; kp < 17; ++kp) { sc[kp] = __expf(sc[kp] - mx); den += sc[kp]; }
    const float inv = 1.f / den;
    for (int e = 0; e < 16; ++e) {
      float o = 0.f;
      for (int kp = 0; kp < 17; ++kp) o += sc[kp] * v[(kp << 8) + hh * 16 + e];
      ctx[((long)b * 17 + qp) * 256 + hh * 16 + e] = o * inv;
    }
  }
}

// out_proj MLP: 256 ->512 ->512 ->512 ->256 on 16-row blocks (WMMA).
__global__ void outproj_kernel(const float* in, OutProjP p, float* out)
{
  extern __shared__ char smem[];
  lch base = (lch)smem;
  lup wt0 = (lup)base;
  lup wt1 = (lup)(base + OFF_WT1);
  lfp red = (lfp)(base + OFF_RED);
  lbf bufA = (lbf)(base + OFF_BUF);
  lbf bufB = (lbf)(base + OFF_BUF + 16 * 512 * 2);
  const int r0 = blockIdx.x * 16;
  load_act(GP(in + (long)r0 * 256), 256, 256, bufA);
  stage_gemm<true, true, false>(bufA, bufB, wt0, wt1, red, GU(p.W1),
      GP(p.b1), GP(p.g1), GP(p.be1), 256, 512, GW(nullptr), 0, 0);
  stage_gemm<true, true, false>(bufB, bufA, wt0, wt1, red, GU(p.W2),
      GP(p.b2), GP(p.g2), GP(p.be2), 512, 512, GW(nullptr), 0, 0);
  stage_gemm<true, true, false>(bufA, bufB, wt0, wt1, red, GU(p.W3),
      GP(p.b3), GP(p.g3), GP(p.be3), 512, 512, GW(nullptr), 0, 0);
  stage_gemm<false, false, true>(bufB, bufA, wt0, wt1, red, GU(p.W4),
      GP(p.b4), GP(nullptr), GP(nullptr), 512, 256, GW(out + (long)r0 * 256), 256, 16);
}

// cross-attn (seq_len 1 -> ctx = LN(vis) @ Wv + bv) + out_proj + classifier.
__global__ void cross_clf_kernel(const float* visattn,
                                 const float* vg_, const float* vb_,
                                 const unsigned int* Wvt, const float* bv_,
                                 OutProjP op, ClfP cp, float* out)
{
  extern __shared__ char smem[];
  lch base = (lch)smem;
  lup wt0 = (lup)base;
  lup wt1 = (lup)(base + OFF_WT1);
  lfp red = (lfp)(base + OFF_RED);
  lbf bufA = (lbf)(base + OFF_BUF);
  lbf bufB = (lbf)(base + OFF_BUF + 16 * 512 * 2);
  const int r0   = blockIdx.x * 16;
  const int tid  = threadIdx.x;
  const int lane = tid & 31;
  const int wv_  = tid >> 5;
  const gfp vg = GP(vg_), vb = GP(vb_);

  for (int m = wv_; m < 16; m += 8) {
    const gfp src = GP(visattn) + (((long)(r0 + m)) * 17 + 8) * 256;  // mid = 8
    float vals[8]; float s = 0.f, qq = 0.f;
#pragma unroll
    for (int j = 0; j < 8; ++j) {
      const int h = lane + 32 * j;
      const float t = src[h];
      vals[j] = t; s += t; qq += t * t;
    }
    for (int xm = 1; xm < 32; xm <<= 1) { s += __shfl_xor(s, xm); qq += __shfl_xor(qq, xm); }
    const float mean = s / 256.f;
    const float rstd = rsqrtf(qq / 256.f - mean * mean + 1e-5f);
#pragma unroll
    for (int j = 0; j < 8; ++j) {
      const int h = lane + 32 * j;
      bufA[m * 256 + h] = f2bf((vals[j] - mean) * rstd * vg[h] + vb[h]);
    }
  }
  __syncthreads();

  stage_gemm<false, false, false>(bufA, bufB, wt0, wt1, red, GU(Wvt),
      GP(bv_), GP(nullptr), GP(nullptr), 256, 256, GW(nullptr), 0, 0);
  stage_gemm<true, true, false>(bufB, bufA, wt0, wt1, red, GU(op.W1),
      GP(op.b1), GP(op.g1), GP(op.be1), 256, 512, GW(nullptr), 0, 0);
  stage_gemm<true, true, false>(bufA, bufB, wt0, wt1, red, GU(op.W2),
      GP(op.b2), GP(op.g2), GP(op.be2), 512, 512, GW(nullptr), 0, 0);
  stage_gemm<true, true, false>(bufB, bufA, wt0, wt1, red, GU(op.W3),
      GP(op.b3), GP(op.g3), GP(op.be3), 512, 512, GW(nullptr), 0, 0);
  stage_gemm<false, false, false>(bufA, bufB, wt0, wt1, red, GU(op.W4),
      GP(op.b4), GP(nullptr), GP(nullptr), 512, 256, GW(nullptr), 0, 0);
  stage_gemm<true, true, false>(bufB, bufA, wt0, wt1, red, GU(cp.W1),
      GP(cp.b1), GP(cp.g1), GP(cp.be1), 256, 256, GW(nullptr), 0, 0);
  stage_gemm<true, true, false>(bufA, bufB, wt0, wt1, red, GU(cp.W2),
      GP(cp.b2), GP(cp.g2), GP(cp.be2), 256, 256, GW(nullptr), 0, 0);
  stage_gemm<true, true, false>(bufB, bufA, wt0, wt1, red, GU(cp.W3),
      GP(cp.b3), GP(cp.g3), GP(cp.be3), 256, 128, GW(nullptr), 0, 0);
  stage_gemm<true, true, false>(bufA, bufB, wt0, wt1, red, GU(cp.W4),
      GP(cp.b4), GP(cp.g4), GP(cp.be4), 128, 64, GW(nullptr), 0, 0);
  stage_gemm<false, false, true>(bufB, bufA, wt0, wt1, red, GU(cp.W5),
      GP(cp.b5), GP(nullptr), GP(nullptr), 64, 2, GW(out + (long)r0 * 2), 2, 16);
}

// ---------------------------------------------------------------------------
// Host launch
// ---------------------------------------------------------------------------
extern "C" void kernel_launch(void* const* d_in, const int* in_sizes, int n_in,
                              void* d_out, int out_size, void* d_ws, size_t ws_size,
                              hipStream_t stream)
{
  (void)in_sizes; (void)n_in; (void)out_size; (void)ws_size;
  auto F = [&](int idx) { return (const float*)d_in[idx]; };

  const float* lang = F(0);
  const float* vis  = F(1);
  // d_in[2] = clip_info (unused by forward)

  // params flattened in setup_inputs() insertion order:
  // lang_head 3..8 | vis_head 9..18 | lang_attn 19..45 | vis_attn 46..72 |
  // cross 73..96 | clf 97..114
  const float *lhW1=F(3), *lhb1=F(4), *lhg1=F(5), *lhbe1=F(6), *lhW2=F(7), *lhb2=F(8);
  const float *vhW1=F(9), *vhb1=F(10), *vhg1=F(11), *vhbe1=F(12),
              *vhW2=F(13), *vhb2=F(14), *vhg2=F(15), *vhbe2=F(16),
              *vhW3=F(17), *vhb3=F(18);
  WsaAttnP la{F(19),F(20),F(21),F(22),F(23),F(24),F(25),F(26),F(27),F(28),F(29),F(30),F(31)};
  WsaAttnP va{F(46),F(47),F(48),F(49),F(50),F(51),F(52),F(53),F(54),F(55),F(56),F(57),F(58)};

  // workspace: six [B,C,256] f32 planes, then packed bf16 weights
  float* ws = (float*)d_ws;
  const long BC = 128L * 17 * 256;
  float* lang_feat = ws + 0 * BC;
  float* vis_feat  = ws + 1 * BC;
  float* lang_ctx  = ws + 2 * BC;
  float* vis_ctx   = ws + 3 * BC;
  float* lang_out  = ws + 4 * BC;
  float* vis_out   = ws + 5 * BC;
  unsigned int* wb = (unsigned int*)(ws + 6 * BC);
  long cur = 0;
  auto packW = [&](const float* Wf, int K, int N, int nclips) {
    unsigned int* dst = wb + cur;
    cur += (long)nclips * wstride(K, N);
    const int grid = nclips * ((N + 127) >> 7) * (K >> 5);
    pack_weight_kernel<<<grid, TB, 0, stream>>>(Wf, dst, K, N);
    return dst;
  };

  // pack all dense-layer weights to bf16 tiles (vis set: 125MB -> L2-resident)
  unsigned int* lhW1t = packW(lhW1, 768, 384, 17);
  unsigned int* lhW2t = packW(lhW2, 384, 256, 17);
  unsigned int* vhW1t = packW(vhW1, 768, 2048, 17);
  unsigned int* vhW2t = packW(vhW2, 2048, 1024, 17);
  unsigned int* vhW3t = packW(vhW3, 1024, 256, 17);
  auto packOut = [&](int b0) {
    OutProjP p;
    p.W1 = packW(F(b0 + 0), 256, 512, 1);  p.b1 = F(b0+1);  p.g1 = F(b0+2);  p.be1 = F(b0+3);
    p.W2 = packW(F(b0 + 4), 512, 512, 1);  p.b2 = F(b0+5);  p.g2 = F(b0+6);  p.be2 = F(b0+7);
    p.W3 = packW(F(b0 + 8), 512, 512, 1);  p.b3 = F(b0+9);  p.g3 = F(b0+10); p.be3 = F(b0+11);
    p.W4 = packW(F(b0 + 12), 512, 256, 1); p.b4 = F(b0+13);
    return p;
  };
  OutProjP laOut = packOut(32);
  OutProjP vaOut = packOut(59);
  const float *crvg = F(75), *crvb = F(76), *crbv = F(82);
  unsigned int* crWvt = packW(F(81), 256, 256, 1);
  OutProjP crOut = packOut(83);
  ClfP clf;
  clf.W1 = packW(F(97), 256, 256, 1);  clf.b1 = F(98);  clf.g1 = F(99);  clf.be1 = F(100);
  clf.W2 = packW(F(101), 256, 256, 1); clf.b2 = F(102); clf.g2 = F(103); clf.be2 = F(104);
  clf.W3 = packW(F(105), 256, 128, 1); clf.b3 = F(106); clf.g3 = F(107); clf.be3 = F(108);
  clf.W4 = packW(F(109), 128, 64, 1);  clf.b4 = F(110); clf.g4 = F(111); clf.be4 = F(112);
  clf.W5 = packW(F(113), 64, 2, 1);    clf.b5 = F(114);

  const int SH_LANG = OFF_BUF + 16 * 768 * 2 + 16 * 384 * 2;   // ~53 KB
  const int SH_VIS  = OFF_BUF + 2 * (16 * 2048 * 2);           // ~145 KB (of 320KB/WGP)
  const int SH_ATTN = 4 * 17 * 256 * 4;                        // ~68 KB
  const int SH_MLP  = OFF_BUF + 2 * (16 * 512 * 2);            // ~49 KB

  (void)hipFuncSetAttribute((const void*)vis_head_kernel,
                            hipFuncAttributeMaxDynamicSharedMemorySize, SH_VIS);
  (void)hipFuncSetAttribute((const void*)wsa_attn_kernel,
                            hipFuncAttributeMaxDynamicSharedMemorySize, SH_ATTN);

  lang_head_kernel<<<8 * 17, TB, SH_LANG, stream>>>(lang, lhW1t, lhb1, lhg1, lhbe1,
                                                    lhW2t, lhb2, lang_feat);
  vis_head_kernel<<<128 * 17, TB, SH_VIS, stream>>>(vis, vhW1t, vhb1, vhg1, vhbe1,
                                                    vhW2t, vhb2, vhg2, vhbe2,
                                                    vhW3t, vhb3, vis_feat);
  wsa_attn_kernel<<<128, TB, SH_ATTN, stream>>>(lang_feat, la, lang_ctx);
  wsa_attn_kernel<<<128, TB, SH_ATTN, stream>>>(vis_feat,  va, vis_ctx);
  outproj_kernel<<<136, TB, SH_MLP, stream>>>(lang_ctx, laOut, lang_out);
  outproj_kernel<<<136, TB, SH_MLP, stream>>>(vis_ctx,  vaOut, vis_out);
  cross_clf_kernel<<<8, TB, SH_MLP, stream>>>(vis_out, crvg, crvb, crWvt, crbv,
                                              crOut, clf, (float*)d_out);
  (void)lang_out; // computed for fidelity with the reference's lang branch
}
